// Classifier_89696097010223
// MI455X (gfx1250) — matrix operand order repaired and checked by reading, not compiled, not run on hardware
//
#include <hip/hip_runtime.h>
#include <hip/hip_bf16.h>

typedef __attribute__((ext_vector_type(16))) _Float16 v16h;
typedef __attribute__((ext_vector_type(8)))  float    v8f;

#define N_NODES  50000
#define N_EDGES  800000
#define FEATS    128
#define HID      64
#define N_GRAPHS 64
#define TYPE_NUM 10

// ---------------------------------------------------------------- zero fill
__global__ void zero_f32(float* __restrict__ p, int n) {
  int i = blockIdx.x * blockDim.x + threadIdx.x;
  if (i < n) p[i] = 0.0f;
}

// ------------------------------------------------- edge gather + scatter-add
// agg[dst] += feat[src];  F/4 lanes per edge, float4 gather + f32 atomics.
// agg is <= 25.6MB -> resident in MI455X's 192MB L2; atomics RMW in L2,
// HBM traffic is the random 256/512B row gathers (bandwidth roofline).
template<int F>
__global__ void scatter_add_kernel(const float* __restrict__ feat,
                                   const int*   __restrict__ src,
                                   const int*   __restrict__ dst,
                                   float*       __restrict__ agg) {
  const int L = F / 4;
  long long tid  = (long long)blockIdx.x * blockDim.x + threadIdx.x;
  int edge = (int)(tid / L);
  int lane = (int)(tid % L);
  if (edge >= N_EDGES) return;
  int s = src[edge];
  int d = dst[edge];
  const float4 v = *(const float4*)(feat + (long long)s * F + lane * 4);
  float* o = agg + (long long)d * F + lane * 4;
  atomicAdd(o + 0, v.x);
  atomicAdd(o + 1, v.y);
  atomicAdd(o + 2, v.z);
  atomicAdd(o + 3, v.w);
}

// ---------------------------------------------------------------- helpers
__device__ __forceinline__ v16h pack16(float4 a0, float4 a1, float4 a2, float4 a3) {
  v16h r;
  r[0]  = (_Float16)a0.x; r[1]  = (_Float16)a0.y; r[2]  = (_Float16)a0.z; r[3]  = (_Float16)a0.w;
  r[4]  = (_Float16)a1.x; r[5]  = (_Float16)a1.y; r[6]  = (_Float16)a1.z; r[7]  = (_Float16)a1.w;
  r[8]  = (_Float16)a2.x; r[9]  = (_Float16)a2.y; r[10] = (_Float16)a2.z; r[11] = (_Float16)a2.w;
  r[12] = (_Float16)a3.x; r[13] = (_Float16)a3.y; r[14] = (_Float16)a3.z; r[15] = (_Float16)a3.w;
  return r;
}

// ---------------------------------------------------------------- WMMA GEMM
// Hout[N,64] = relu(Aagg[N,K] @ Wrel[K,64] + Aroot[N,K] @ Wroot[K,64])
// Weights are swizzled into the exact WMMA B-fragment layout in LDS:
// fragment (k-chunk, n-tile) = 32 lanes x 16 contiguous halves (1KB), so a
// per-lane fragment fetch is one aligned 32B read -> 2x ds_load_b128 instead
// of 16x ds_load_u16. 50000 = 3125*16, so every wave owns whole 16-row strips
// and EXEC is all-ones around each v_wmma (ISA requirement).
template<int K>
__global__ void __launch_bounds__(128)
gemm_relu_wmma(const float* __restrict__ Aagg,
               const float* __restrict__ Aroot,
               const float* __restrict__ Wrel,
               const float* __restrict__ Wroot,
               float*       __restrict__ Hout) {
  constexpr int NCHUNK = K / 32;
  constexpr int NFRAG  = NCHUNK * 4;          // fragments per matrix
  __shared__ _Float16 sW[2 * NFRAG * 512];    // K=128 -> 32KB, K=64 -> 16KB
  const int tid = threadIdx.x;

  // Swizzle fill: element W[k][c] -> fragment(chunk=k/32, nt=c/16),
  // lane = ((k%32)/8%2)*16 + c%16, half j = k%8 + (k%32)/16*8.
  for (int i = tid; i < K * HID; i += 128) {
    const int k     = i >> 6;                 // HID = 64
    const int c     = i & 63;
    const int chunk = k >> 5;
    const int kk    = k & 31;
    const int j     = (kk & 7) | ((kk >> 4) << 3);
    const int hi    = (kk >> 3) & 1;
    const int lane  = hi * 16 + (c & 15);
    const int nt    = c >> 4;
    const int dst   = ((chunk * 4 + nt) << 9) + (lane << 4) + j;
    sW[dst]               = (_Float16)Wrel[i];
    sW[NFRAG * 512 + dst] = (_Float16)Wroot[i];
  }
  __syncthreads();

  const int lane   = tid & 31;
  const int gwave  = blockIdx.x * 4 + (tid >> 5);
  const int nwaves = gridDim.x * 4;
  const int m      = lane & 15;               // A row-in-tile / C col-in-tile
  const int hiHalf = lane >> 4;
  const int kbA    = hiHalf * 8;              // A fragment K base
  const v16h* bp   = (const v16h*)sW;         // 32 v16h per fragment

  for (int strip = gwave; strip < N_NODES / 16; strip += nwaves) {
    const int rowBase = strip * 16 + m;
    v8f acc[4] = {};

#pragma unroll
    for (int chunk = 0; chunk < NCHUNK; ++chunk) {
      const int off = rowBase * K + chunk * 32 + kbA;
      const float4* pA4 = (const float4*)(Aagg  + off);
      const float4* pR4 = (const float4*)(Aroot + off);
      // lane holds K = {0..7, 16..23} (+8 for hi half): two 8-float runs
      const v16h aA = pack16(pA4[0], pA4[1], pA4[4], pA4[5]);
      const v16h aR = pack16(pR4[0], pR4[1], pR4[4], pR4[5]);

#pragma unroll
      for (int nt = 0; nt < 4; ++nt) {
        const v16h bRel  = bp[((chunk * 4 + nt) << 5) + lane];
        const v16h bRoot = bp[(NFRAG << 5) + ((chunk * 4 + nt) << 5) + lane];
        acc[nt] = __builtin_amdgcn_wmma_f32_16x16x32_f16(
            false, aA, false, bRel, (short)0, acc[nt], false, false);
        acc[nt] = __builtin_amdgcn_wmma_f32_16x16x32_f16(
            false, aR, false, bRoot, (short)0, acc[nt], false, false);
      }
    }

    // C/D layout: lane n<16 -> col n rows 0..7; lanes 16..31 -> rows 8..15
#pragma unroll
    for (int nt = 0; nt < 4; ++nt) {
      const int col = nt * 16 + m;
#pragma unroll
      for (int m8 = 0; m8 < 8; ++m8) {
        const int r = strip * 16 + hiHalf * 8 + m8;
        const float v = acc[nt][m8];
        Hout[r * HID + col] = v > 0.0f ? v : 0.0f;
      }
    }
  }
}

// ------------------------------------------------------------- mean pooling
__global__ void pool_kernel(const float* __restrict__ h,
                            const int*   __restrict__ batch,
                            float* __restrict__ sums,
                            float* __restrict__ counts) {
  const int node = blockIdx.x;
  const int t    = threadIdx.x;   // 64 threads = HID
  const int g    = batch[node];
  atomicAdd(&sums[g * HID + t], h[node * HID + t]);
  if (t == 0) atomicAdd(&counts[g], 1.0f);
}

// --------------------------------------------------- head: fc + softmax[0]
__global__ void head_kernel(const float* __restrict__ sums,
                            const float* __restrict__ counts,
                            const float* __restrict__ Wfc,
                            const float* __restrict__ bfc,
                            float* __restrict__ out) {
  __shared__ float logits[TYPE_NUM];
  const int t = threadIdx.x;
  float c = counts[0];
  c = (c < 1.0f) ? 1.0f : c;
  if (t < TYPE_NUM) {
    float acc = bfc[t];
    for (int hh = 0; hh < HID; ++hh)
      acc += (sums[hh] / c) * Wfc[hh * TYPE_NUM + t];
    logits[t] = acc;
  }
  __syncthreads();
  if (t == 0) {
    float mx = logits[0];
    for (int i = 1; i < TYPE_NUM; ++i) mx = fmaxf(mx, logits[i]);
    float e[TYPE_NUM];
    float s = 0.0f;
    for (int i = 0; i < TYPE_NUM; ++i) { e[i] = __expf(logits[i] - mx); s += e[i]; }
    for (int i = 0; i < TYPE_NUM; ++i) out[i] = e[i] / s;
  }
}

// -------------------------------------------------------------------- launch
extern "C" void kernel_launch(void* const* d_in, const int* in_sizes, int n_in,
                              void* d_out, int out_size, void* d_ws, size_t ws_size,
                              hipStream_t stream) {
  const float* x      = (const float*)d_in[0];
  const int*   edge   = (const int*)d_in[1];
  const int*   src    = edge;
  const int*   dst    = edge + N_EDGES;
  const int*   batch  = (const int*)d_in[2];
  const float* Wrel1  = (const float*)d_in[3];
  const float* Wroot1 = (const float*)d_in[4];
  const float* Wrel2  = (const float*)d_in[5];
  const float* Wroot2 = (const float*)d_in[6];
  const float* Wrel3  = (const float*)d_in[7];
  const float* Wroot3 = (const float*)d_in[8];
  const float* Wfc    = (const float*)d_in[9];
  const float* bfc    = (const float*)d_in[10];
  float* out = (float*)d_out;

  // workspace carve-out (all f32):
  float* agg    = (float*)d_ws;                          // 50000*128 (reused as 50000*64)
  float* hA     = agg + (size_t)N_NODES * FEATS;         // 50000*64
  float* hB     = hA  + (size_t)N_NODES * HID;           // 50000*64
  float* sums   = hB  + (size_t)N_NODES * HID;           // 64*64
  float* counts = sums + (size_t)N_GRAPHS * HID;         // 64

  const int ZB = 256;
  const int GEMM_BLOCKS = 256;   // 1024 waves grid-striding 3125 strips

  // ---- layer 1 (K = 128)
  zero_f32<<<(N_NODES * FEATS + ZB - 1) / ZB, ZB, 0, stream>>>(agg, N_NODES * FEATS);
  scatter_add_kernel<FEATS><<<(N_EDGES * (FEATS / 4) + 255) / 256, 256, 0, stream>>>(x, src, dst, agg);
  gemm_relu_wmma<FEATS><<<GEMM_BLOCKS, 128, 0, stream>>>(agg, x, Wrel1, Wroot1, hA);

  // ---- layer 2 (K = 64)
  zero_f32<<<(N_NODES * HID + ZB - 1) / ZB, ZB, 0, stream>>>(agg, N_NODES * HID);
  scatter_add_kernel<HID><<<(N_EDGES * (HID / 4) + 255) / 256, 256, 0, stream>>>(hA, src, dst, agg);
  gemm_relu_wmma<HID><<<GEMM_BLOCKS, 128, 0, stream>>>(agg, hA, Wrel2, Wroot2, hB);

  // ---- layer 3 (K = 64)
  zero_f32<<<(N_NODES * HID + ZB - 1) / ZB, ZB, 0, stream>>>(agg, N_NODES * HID);
  scatter_add_kernel<HID><<<(N_EDGES * (HID / 4) + 255) / 256, 256, 0, stream>>>(hB, src, dst, agg);
  gemm_relu_wmma<HID><<<GEMM_BLOCKS, 128, 0, stream>>>(agg, hB, Wrel3, Wroot3, hA);

  // ---- global mean pool + head (output = softmax of graph 0 only)
  zero_f32<<<(N_GRAPHS * HID + N_GRAPHS + ZB - 1) / ZB, ZB, 0, stream>>>(sums, N_GRAPHS * HID + N_GRAPHS);
  pool_kernel<<<N_NODES, HID, 0, stream>>>(hA, batch, sums, counts);
  head_kernel<<<1, 32, 0, stream>>>(sums, counts, Wfc, bfc, out);
}
